// Readout_30365418783528
// MI455X (gfx1250) — compile-verified, matrix operand order checked by reading, not run on hardware
//
#include <hip/hip_runtime.h>

#define NV        1000000
#define NGRAPH    50000
#define NDIM      128
#define HID       128
#define ODIM      128
#define GDIM      64

typedef __attribute__((ext_vector_type(16))) __bf16        v16bf;
typedef __attribute__((ext_vector_type(2)))  __bf16        v2bf;
typedef __attribute__((ext_vector_type(8)))  float         v8f;
typedef __attribute__((ext_vector_type(8)))  unsigned int  v8u;

__device__ __forceinline__ unsigned short f32_to_bf16_bits(float f) {
  unsigned u = __builtin_bit_cast(unsigned, f);
  unsigned r = u + 0x7FFFu + ((u >> 16) & 1u);   // round-to-nearest-even
  return (unsigned short)(r >> 16);
}
__device__ __forceinline__ float bf16_bits_to_f32(unsigned short s) {
  unsigned u = ((unsigned)s) << 16;
  return __builtin_bit_cast(float, u);
}

// Split two f32 into packed-bf16 hi dword + packed-bf16 lo (residual) dword.
// Prefer the single-op hardware packed converter V_CVT_PK_BF16_F32 (RNE).
__device__ __forceinline__ void cvt_pair(float a, float b,
                                         unsigned& hi, unsigned& lo) {
#if __has_builtin(__builtin_amdgcn_cvt_pk_bf16_f32)
  v2bf h = __builtin_amdgcn_cvt_pk_bf16_f32(a, b);
  unsigned hbits = __builtin_bit_cast(unsigned, h);
  float ha = __builtin_bit_cast(float, hbits << 16);          // elem a as f32
  float hb = __builtin_bit_cast(float, hbits & 0xFFFF0000u);  // elem b as f32
  v2bf l = __builtin_amdgcn_cvt_pk_bf16_f32(a - ha, b - hb);
  hi = hbits;
  lo = __builtin_bit_cast(unsigned, l);
#else
  unsigned short ha = f32_to_bf16_bits(a), hb = f32_to_bf16_bits(b);
  unsigned short la = f32_to_bf16_bits(a - bf16_bits_to_f32(ha));
  unsigned short lb = f32_to_bf16_bits(b - bf16_bits_to_f32(hb));
  hi = (unsigned)ha | ((unsigned)hb << 16);
  lo = (unsigned)la | ((unsigned)lb << 16);
#endif
}

__device__ __forceinline__ v8f wmma_bf16(v16bf a, v16bf b, v8f c) {
  return __builtin_amdgcn_wmma_f32_16x16x32_bf16(false, a, false, b, (short)0, c,
                                                 false, false);
}

// Fragment layout in workspace: [frag][lane][8 dwords] -> one 32B load per lane.
__device__ __forceinline__ v16bf load_frag(const unsigned* __restrict__ base,
                                           int frag, int lane) {
  v8u q = *reinterpret_cast<const v8u*>(base + frag * 256 + lane * 8);
  return __builtin_bit_cast(v16bf, q);
}

// Split 16 f32 (two 8-float runs) into bf16 hi + bf16 lo fragments.
__device__ __forceinline__ void cvt_split(v8f f0, v8f f1, v16bf& hi, v16bf& lo) {
  v8u H, L;
#pragma unroll
  for (int p = 0; p < 4; ++p) {
    unsigned h, lw;
    cvt_pair(f0[2 * p], f0[2 * p + 1], h, lw);
    H[p] = h;
    L[p] = lw;
  }
#pragma unroll
  for (int p = 0; p < 4; ++p) {
    unsigned h, lw;
    cvt_pair(f1[2 * p], f1[2 * p + 1], h, lw);
    H[4 + p] = h;
    L[4 + p] = lw;
  }
  hi = __builtin_bit_cast(v16bf, H);
  lo = __builtin_bit_cast(v16bf, L);
}

// ---------------------------------------------------------------------------
// Zero the Z1 + counts accumulators (must run every launch; ws is not cleared).
__global__ void zero_ws_kernel(float* __restrict__ p, int n) {
  int i = blockIdx.x * blockDim.x + threadIdx.x;
  if (i < n) p[i] = 0.0f;
}

// ---------------------------------------------------------------------------
// Pre-swizzle W1/W2 (row-major [128][128], used as B with N=row, K=col) into
// CDNA5 16x16x32 bf16 B-operand fragments, split hi/lo.
__global__ void prep_w12_kernel(const float* __restrict__ W1,
                                const float* __restrict__ W2,
                                unsigned* __restrict__ w1hi, unsigned* __restrict__ w1lo,
                                unsigned* __restrict__ w2hi, unsigned* __restrict__ w2lo) {
  int tid = blockIdx.x * blockDim.x + threadIdx.x;
  if (tid >= 32 * 32) return;                  // 8 col-tiles * 4 k-tiles, 32 lanes
  int frag = tid >> 5, lane = tid & 31;
  int ct = frag >> 2, kt = frag & 3;
  int n = (lane & 15) + 16 * ct, hsel = lane >> 4;
#pragma unroll
  for (int v = 0; v < 8; ++v) {
    int k0  = kt * 32 + hsel * 8 + (v < 4 ? 2 * v : 16 + 2 * (v - 4));
    int idx = frag * 256 + lane * 8 + v;
    cvt_pair(W1[n * NDIM + k0], W1[n * NDIM + k0 + 1], w1hi[idx], w1lo[idx]);
    cvt_pair(W2[n * NDIM + k0], W2[n * NDIM + k0 + 1], w2hi[idx], w2lo[idx]);
  }
}

// W3: row-major [128][320] -> 8 col-tiles * 10 k-tiles fragments.
__global__ void prep_w3_kernel(const float* __restrict__ W3,
                               unsigned* __restrict__ w3hi, unsigned* __restrict__ w3lo) {
  int tid = blockIdx.x * blockDim.x + threadIdx.x;
  if (tid >= 80 * 32) return;
  int frag = tid >> 5, lane = tid & 31;
  int ct = frag / 10, kt = frag % 10;
  int n = (lane & 15) + 16 * ct, hsel = lane >> 4;
#pragma unroll
  for (int v = 0; v < 8; ++v) {
    int k0  = kt * 32 + hsel * 8 + (v < 4 ? 2 * v : 16 + 2 * (v - 4));
    int idx = frag * 256 + lane * 8 + v;
    cvt_pair(W3[n * (2 * HID + GDIM) + k0], W3[n * (2 * HID + GDIM) + k0 + 1],
             w3hi[idx], w3lo[idx]);
  }
}

// ---------------------------------------------------------------------------
// Main: gated MLP via split-bf16 WMMA + atomic segment-sum. One wave = 16 nodes.
__global__ void __launch_bounds__(256)
gated_segsum_kernel(const float* __restrict__ x, const int* __restrict__ n2g,
                    const float* __restrict__ b1, const float* __restrict__ b2,
                    const unsigned* __restrict__ w1hi, const unsigned* __restrict__ w1lo,
                    const unsigned* __restrict__ w2hi, const unsigned* __restrict__ w2lo,
                    float* __restrict__ Z1, float* __restrict__ counts) {
  const int wavesPerBlock = blockDim.x >> 5;
  int wave = blockIdx.x * wavesPerBlock + (threadIdx.x >> 5);
  if (wave >= NV / 16) return;                 // wave-uniform: EXEC stays all-ones
  int lane = threadIdx.x & 31;
  int l = lane & 15, hsel = lane >> 4;
  int rowBase = wave * 16;

  // ---- A operand: this lane's row (M = l), K-halves by hsel, 4 k-tiles ----
  v16bf Ahi[4], Alo[4];
  const float* xrow = x + (size_t)(rowBase + l) * NDIM;
#pragma unroll
  for (int kt = 0; kt < 4; ++kt) {
    v8f f0 = *reinterpret_cast<const v8f*>(xrow + kt * 32 + hsel * 8);
    v8f f1 = *reinterpret_cast<const v8f*>(xrow + kt * 32 + 16 + hsel * 8);
    cvt_split(f0, f1, Ahi[kt], Alo[kt]);
  }

  // Graph ids for the 8 D-rows this lane owns (M = r + 8*hsel).
  int gidx[8];
#pragma unroll
  for (int r = 0; r < 8; ++r) gidx[r] = n2g[rowBase + r + 8 * hsel];

  // One count contribution per node (lanes 0..15 cover the 16 rows).
  if (lane < 16) atomicAdd(&counts[n2g[rowBase + l]], 1.0f);

  for (int ct = 0; ct < 8; ++ct) {
    v8f acc1 = {};
    v8f acc2 = {};
#pragma unroll
    for (int kt = 0; kt < 4; ++kt) {
      int fr = ct * 4 + kt;
      v16bf B1h = load_frag(w1hi, fr, lane);
      v16bf B1l = load_frag(w1lo, fr, lane);
      v16bf B2h = load_frag(w2hi, fr, lane);
      v16bf B2l = load_frag(w2lo, fr, lane);
      acc1 = wmma_bf16(Ahi[kt], B1h, acc1);
      acc1 = wmma_bf16(Ahi[kt], B1l, acc1);
      acc1 = wmma_bf16(Alo[kt], B1h, acc1);
      acc2 = wmma_bf16(Ahi[kt], B2h, acc2);
      acc2 = wmma_bf16(Ahi[kt], B2l, acc2);
      acc2 = wmma_bf16(Alo[kt], B2h, acc2);
    }
    int n = l + 16 * ct;
    float bb1 = b1[n], bb2 = b2[n];
#pragma unroll
    for (int r = 0; r < 8; ++r) {
      float z1v = acc1[r] + bb1;
      float z2v = acc2[r] + bb2;
      float hv  = z1v * (1.0f / (1.0f + __expf(-z2v)));
      atomicAdd(&Z1[(size_t)gidx[r] * HID + n], hv);
    }
  }
}

// ---------------------------------------------------------------------------
// Readout: out = relu([Z1 | Z1/max(c,1) | gx] @ W3^T + b3). One wave = 16 graphs.
__global__ void __launch_bounds__(256)
readout_kernel(const float* __restrict__ Z1, const float* __restrict__ counts,
               const float* __restrict__ gx, const float* __restrict__ b3,
               const unsigned* __restrict__ w3hi, const unsigned* __restrict__ w3lo,
               float* __restrict__ out) {
  const int wavesPerBlock = blockDim.x >> 5;
  int wave = blockIdx.x * wavesPerBlock + (threadIdx.x >> 5);
  if (wave >= NGRAPH / 16) return;
  int lane = threadIdx.x & 31;
  int l = lane & 15, hsel = lane >> 4;
  int gBase = wave * 16;
  int g = gBase + l;

  float rc = 1.0f / fmaxf(counts[g], 1.0f);
  const float* z1row = Z1 + (size_t)g * HID;
  const float* gxrow = gx + (size_t)g * GDIM;

  // Build the 16x320 A operand (10 k-tiles); each 8-float run stays in-region.
  v16bf Ahi[10], Alo[10];
#pragma unroll
  for (int kt = 0; kt < 10; ++kt) {
    const float* base;
    float sc;
    if (kt < 4)      { base = z1row + kt * 32;       sc = 1.0f; }
    else if (kt < 8) { base = z1row + (kt - 4) * 32; sc = rc;   }
    else             { base = gxrow + (kt - 8) * 32; sc = 1.0f; }
    v8f f0 = *reinterpret_cast<const v8f*>(base + hsel * 8);
    v8f f1 = *reinterpret_cast<const v8f*>(base + 16 + hsel * 8);
    f0 = f0 * sc;
    f1 = f1 * sc;
    cvt_split(f0, f1, Ahi[kt], Alo[kt]);
  }

  for (int ct = 0; ct < 8; ++ct) {
    v8f acc = {};
#pragma unroll
    for (int kt = 0; kt < 10; ++kt) {
      int fr = ct * 10 + kt;
      v16bf Bh = load_frag(w3hi, fr, lane);
      v16bf Bl = load_frag(w3lo, fr, lane);
      acc = wmma_bf16(Ahi[kt], Bh, acc);
      acc = wmma_bf16(Ahi[kt], Bl, acc);
      acc = wmma_bf16(Alo[kt], Bh, acc);
    }
    int n = l + 16 * ct;
    float bb = b3[n];
#pragma unroll
    for (int r = 0; r < 8; ++r) {
      out[(size_t)(gBase + r + 8 * hsel) * ODIM + n] = fmaxf(acc[r] + bb, 0.0f);
    }
  }
}

// ---------------------------------------------------------------------------
extern "C" void kernel_launch(void* const* d_in, const int* in_sizes, int n_in,
                              void* d_out, int out_size, void* d_ws, size_t ws_size,
                              hipStream_t stream) {
  (void)in_sizes; (void)n_in; (void)out_size; (void)ws_size;
  const float* x   = (const float*)d_in[0];
  const int*   n2g = (const int*)d_in[1];
  const float* gx  = (const float*)d_in[2];
  const float* W1  = (const float*)d_in[3];
  const float* b1  = (const float*)d_in[4];
  const float* W2  = (const float*)d_in[5];
  const float* b2  = (const float*)d_in[6];
  const float* W3  = (const float*)d_in[7];
  const float* b3  = (const float*)d_in[8];
  float* out = (float*)d_out;

  // Workspace layout (all 16B-aligned): Z1 | counts | weight fragments.
  float* Z1     = (float*)d_ws;                          // 50000*128 floats
  float* counts = Z1 + (size_t)NGRAPH * HID;             // 50000 floats
  unsigned* w1hi = (unsigned*)(counts + NGRAPH);
  unsigned* w1lo = w1hi + 32 * 256;
  unsigned* w2hi = w1lo + 32 * 256;
  unsigned* w2lo = w2hi + 32 * 256;
  unsigned* w3hi = w2lo + 32 * 256;
  unsigned* w3lo = w3hi + 80 * 256;

  const int nzero = NGRAPH * HID + NGRAPH;               // Z1 + counts contiguous
  zero_ws_kernel<<<(nzero + 255) / 256, 256, 0, stream>>>(Z1, nzero);
  prep_w12_kernel<<<4, 256, 0, stream>>>(W1, W2, w1hi, w1lo, w2hi, w2lo);
  prep_w3_kernel<<<10, 256, 0, stream>>>(W3, w3hi, w3lo);

  const int nTiles = NV / 16;                            // 62500 waves
  gated_segsum_kernel<<<(nTiles + 7) / 8, 256, 0, stream>>>(
      x, n2g, b1, b2, w1hi, w1lo, w2hi, w2lo, Z1, counts);

  const int gTiles = NGRAPH / 16;                        // 3125 waves
  readout_kernel<<<(gTiles + 7) / 8, 256, 0, stream>>>(
      Z1, counts, gx, b3, w3hi, w3lo, out);
}
